// QANet_57939108823270
// MI455X (gfx1250) — matrix operand order repaired and checked by reading, not compiled, not run on hardware
//
#include <hip/hip_runtime.h>
#include <hip/hip_bf16.h>
#include <hip/hip_fp16.h>

typedef __attribute__((ext_vector_type(16))) _Float16 v16h;
typedef __attribute__((ext_vector_type(8)))  _Float16 v8h;
typedef __attribute__((ext_vector_type(8)))  float    v8f;

#define BATCH 32
#define LCTX  2048
#define LQRY  128
#define DDIM  128
#define NIT   (LCTX / 128)   // 16 context row-tiles

static __device__ __forceinline__ v8f wmma_f16(v16h a, v16h b, v8f c) {
  return __builtin_amdgcn_wmma_f32_16x16x32_f16(false, a, false, b, (short)0, c,
                                                false, false);
}

// A-fragment loader (16x32 f16): p = row + 32*ks + 16*h; elements
// p[0..15] (contiguous k = 16h + s, matching the B loads).
static __device__ __forceinline__ v16h load_afrag(const _Float16* p) {
  v8h lo = *(const v8h*)(p);
  v8h hi = *(const v8h*)(p + 8);
  return __builtin_shufflevector(lo, hi, 0,1,2,3,4,5,6,7,8,9,10,11,12,13,14,15);
}

// Async copy of one 16B chunk from global to LDS (CDNA5 ASYNCcnt path).
static __device__ __forceinline__ void async_g2l_b128(unsigned lds_off,
                                                      const void* gptr) {
  unsigned long long ga = (unsigned long long)gptr;
  asm volatile("global_load_async_to_lds_b128 %0, %1, off"
               :: "v"(lds_off), "v"(ga) : "memory");
}
static __device__ __forceinline__ void wait_async0() {
  asm volatile("s_wait_asynccnt 0x0" ::: "memory");
}

#define PADW 132  // padded f32 row width in LDS (16B-multiple, breaks bank stride)

// ---------------------------------------------------------------------------
// Kernel 0a: per (b, 128-row C tile), staged via async-LDS in 64-row halves:
//   Cw3h = f16(C * w3) row-major;  ChT = f16(C) transposed (b,d,i);
//   c1[b,i] = C[b,i,:] @ w1
// ---------------------------------------------------------------------------
__global__ __launch_bounds__(256) void k0_prepC(
    const float* __restrict__ C, const float* __restrict__ w,
    _Float16* __restrict__ Cw3h, _Float16* __restrict__ ChT,
    float* __restrict__ c1g)
{
  const int b = blockIdx.y, it = blockIdx.x;
  const int i0 = it * 128;
  const int tid = threadIdx.x;

  __shared__ float sT[64 * PADW];  // 33.8 KB
  const unsigned lbase = (unsigned)(size_t)(&sT[0]);

  for (int half = 0; half < 2; half++) {
    const int r0 = i0 + 64 * half;
    const float* csrc = C + ((size_t)b * LCTX + r0) * DDIM;
    // 64 rows * 512B = 2048 chunks of 16B; 8 per thread
#pragma unroll
    for (int r = 0; r < 8; r++) {
      const int chunk = tid + 256 * r;
      const int row = chunk >> 5;           // 32 chunks per 512B row
      const int cb  = (chunk & 31) * 16;    // byte within row
      async_g2l_b128(lbase + (unsigned)(row * PADW * 4 + cb),
                     (const void*)(csrc + chunk * 4));
    }
    wait_async0();
    __syncthreads();

    // c1 row dots (64 rows)
    if (tid < 64) {
      const float* row = &sT[tid * PADW];
      float a = 0.f;
      for (int k = 0; k < 128; k++) a += row[k] * w[k];
      c1g[(size_t)b * LCTX + r0 + tid] = a;
    }
    // Cw3h (row-major): 8192 elements, 32 per thread
    _Float16* cw = Cw3h + ((size_t)b * LCTX + r0) * DDIM;
#pragma unroll
    for (int r = 0; r < 32; r++) {
      const int idx = tid + 256 * r;
      const int i = idx >> 7, d = idx & 127;
      cw[idx] = (_Float16)(sT[i * PADW + d] * w[256 + d]);
    }
    // ChT (transposed): thread t -> column d = t>>1, 32 i's
    {
      const int d = tid >> 1, hf = tid & 1;
      _Float16* ct = ChT + ((size_t)b * DDIM + d) * LCTX + r0 + hf * 32;
      for (int ii = 0; ii < 32; ii++)
        ct[ii] = (_Float16)sT[(hf * 32 + ii) * PADW + d];
    }
    __syncthreads();
  }
}

// ---------------------------------------------------------------------------
// Kernel 0b: per batch: Qh = f16(Q) row-major; QhT transposed; q2 = Q @ w2
// ---------------------------------------------------------------------------
__global__ __launch_bounds__(256) void k0_prepQ(
    const float* __restrict__ Q, const float* __restrict__ w,
    _Float16* __restrict__ Qh, _Float16* __restrict__ QhT,
    float* __restrict__ q2g)
{
  const int b = blockIdx.x;
  const int tid = threadIdx.x;

  __shared__ float sT[64 * PADW];
  const unsigned lbase = (unsigned)(size_t)(&sT[0]);

  for (int half = 0; half < 2; half++) {
    const int r0 = 64 * half;
    const float* qsrc = Q + ((size_t)b * LQRY + r0) * DDIM;
#pragma unroll
    for (int r = 0; r < 8; r++) {
      const int chunk = tid + 256 * r;
      const int row = chunk >> 5;
      const int cb  = (chunk & 31) * 16;
      async_g2l_b128(lbase + (unsigned)(row * PADW * 4 + cb),
                     (const void*)(qsrc + chunk * 4));
    }
    wait_async0();
    __syncthreads();

    if (tid < 64) {
      const float* row = &sT[tid * PADW];
      float a = 0.f;
      for (int k = 0; k < 128; k++) a += row[k] * w[128 + k];
      q2g[(size_t)b * LQRY + r0 + tid] = a;
    }
    _Float16* qh = Qh + ((size_t)b * LQRY + r0) * DDIM;
#pragma unroll
    for (int r = 0; r < 32; r++) {
      const int idx = tid + 256 * r;
      const int j = idx >> 7, d = idx & 127;
      qh[idx] = (_Float16)sT[j * PADW + d];
    }
    {
      const int d = tid >> 1, hf = tid & 1;
      _Float16* qt = QhT + ((size_t)b * DDIM + d) * LQRY + r0 + hf * 32;
      for (int jj = 0; jj < 32; jj++)
        qt[jj] = (_Float16)sT[(hf * 32 + jj) * PADW + d];
    }
    __syncthreads();
  }
}

// ---------------------------------------------------------------------------
// Kernel 1: per (b, 128-row tile): S = Cw3h @ Qh^T + c1 + q2 (WMMA);
//   S1 = softmax_j (masked) -> S1h f16;  P2 = exp(S2log - tile colmax)
//   stored transposed f16 (b,j,i); TMAX/TSUM per-tile column stats.
// ---------------------------------------------------------------------------
__global__ __launch_bounds__(256) void k1_score(
    const _Float16* __restrict__ Cw3h, const _Float16* __restrict__ Qh,
    const float* __restrict__ c1g, const float* __restrict__ q2g,
    const float* __restrict__ cmask, const float* __restrict__ qmask,
    _Float16* __restrict__ S1h, _Float16* __restrict__ P2T,
    float* __restrict__ TMAX, float* __restrict__ TSUM)
{
  const int b = blockIdx.y, it = blockIdx.x;
  const int i0 = it * 128;
  const int tid = threadIdx.x, lane = tid & 31, wv = tid >> 5;
  const int h = lane >> 4, n = lane & 15, m = lane & 15;

  __shared__ float sc1[128], sq2[128], sqm[128], scm[128];
  __shared__ float sred[8][128];
  __shared__ float stmax[128];

  if (tid < 128) {
    sc1[tid] = c1g[(size_t)b * LCTX + i0 + tid];
    scm[tid] = cmask[(size_t)b * LCTX + i0 + tid];
    sq2[tid] = q2g[(size_t)b * LQRY + tid];
    sqm[tid] = qmask[(size_t)b * LQRY + tid];
  }
  __syncthreads();

  // A fragments: rows of Cw3h (pure f16 loads)
  const _Float16* crow = Cw3h + ((size_t)b * LCTX + i0 + 16 * wv + m) * DDIM;
  v16h afr[4];
#pragma unroll
  for (int ks = 0; ks < 4; ks++) afr[ks] = load_afrag(crow + 32 * ks + 16 * h);

  v8f acc[8];
#pragma unroll
  for (int nt = 0; nt < 8; nt++) {
    v8f c = {};
    const _Float16* qrow = Qh + ((size_t)b * LQRY + 16 * nt + n) * DDIM;
#pragma unroll
    for (int ks = 0; ks < 4; ks++) {
      v16h bf = *(const v16h*)(qrow + 32 * ks + 16 * h);  // B[k=d][n=j]=Q[j][d]
      c = wmma_f16(afr[ks], bf, c);
    }
    acc[nt] = c;
  }

#pragma unroll
  for (int nt = 0; nt < 8; nt++)
#pragma unroll
    for (int v = 0; v < 8; v++)
      acc[nt][v] += sc1[16 * wv + v + 8 * h] + sq2[16 * nt + n];

  // ---- S1: masked row softmax over j ----
#pragma unroll
  for (int v = 0; v < 8; v++) {
    float mx = -INFINITY;
#pragma unroll
    for (int nt = 0; nt < 8; nt++) {
      float qm = sqm[16 * nt + n];
      float l  = acc[nt][v] * (1.f - qm) + qm * (-1e30f);
      mx = fmaxf(mx, l);
    }
    mx = fmaxf(mx, __shfl_xor(mx, 1, 32));
    mx = fmaxf(mx, __shfl_xor(mx, 2, 32));
    mx = fmaxf(mx, __shfl_xor(mx, 4, 32));
    mx = fmaxf(mx, __shfl_xor(mx, 8, 32));
    float p[8], sum = 0.f;
#pragma unroll
    for (int nt = 0; nt < 8; nt++) {
      float qm = sqm[16 * nt + n];
      float l  = acc[nt][v] * (1.f - qm) + qm * (-1e30f);
      p[nt] = __expf(l - mx);
      sum += p[nt];
    }
    sum += __shfl_xor(sum, 1, 32);
    sum += __shfl_xor(sum, 2, 32);
    sum += __shfl_xor(sum, 4, 32);
    sum += __shfl_xor(sum, 8, 32);
    const float inv = 1.f / sum;
    const size_t rb = ((size_t)b * LCTX + i0 + 16 * wv + v + 8 * h) * LQRY;
#pragma unroll
    for (int nt = 0; nt < 8; nt++)
      S1h[rb + 16 * nt + n] = (_Float16)(p[nt] * inv);
  }

  // ---- S2: per-tile column max ----
#pragma unroll
  for (int nt = 0; nt < 8; nt++) {
    float cmx = -INFINITY;
#pragma unroll
    for (int v = 0; v < 8; v++) {
      float cm = scm[16 * wv + v + 8 * h];
      float l  = acc[nt][v] * (1.f - cm) + cm * (-1e30f);
      cmx = fmaxf(cmx, l);
    }
    cmx = fmaxf(cmx, __shfl_xor(cmx, 16, 32));
    if (h == 0) sred[wv][16 * nt + n] = cmx;
  }
  __syncthreads();
  if (tid < 128) {
    float mx = -INFINITY;
    for (int ww = 0; ww < 8; ww++) mx = fmaxf(mx, sred[ww][tid]);
    stmax[tid] = mx;
  }
  __syncthreads();

  // ---- exp vs tile max, column sums, transposed f16 store ----
#pragma unroll
  for (int nt = 0; nt < 8; nt++) {
    const int j = 16 * nt + n;
    const float tmx = stmax[j];
    float ssum = 0.f;
    _Float16* pb = P2T + ((size_t)b * LQRY + j) * LCTX + i0 + 16 * wv + 8 * h;
#pragma unroll
    for (int v = 0; v < 8; v++) {
      float cm = scm[16 * wv + v + 8 * h];
      float l  = acc[nt][v] * (1.f - cm) + cm * (-1e30f);
      float p2 = __expf(l - tmx);
      ssum += p2;
      pb[v] = (_Float16)p2;
    }
    ssum += __shfl_xor(ssum, 16, 32);
    if (h == 0) sred[wv][j] = ssum;
  }
  __syncthreads();
  if (tid < 128) {
    float s = 0.f;
    for (int ww = 0; ww < 8; ww++) s += sred[ww][tid];
    TMAX[((size_t)b * NIT + it) * 128 + tid] = stmax[tid];
    TSUM[((size_t)b * NIT + it) * 128 + tid] = s;
  }
}

// ---------------------------------------------------------------------------
// Kernel 2: SCL[b,t,j] = exp(TMAX - gmax) / gsum
// ---------------------------------------------------------------------------
__global__ __launch_bounds__(128) void k2_colstats(
    const float* __restrict__ TMAX, const float* __restrict__ TSUM,
    float* __restrict__ SCL)
{
  const int b = blockIdx.x, j = threadIdx.x;
  float tm[NIT], gm = -INFINITY;
#pragma unroll
  for (int t = 0; t < NIT; t++) {
    tm[t] = TMAX[((size_t)b * NIT + t) * 128 + j];
    gm = fmaxf(gm, tm[t]);
  }
  float gs = 0.f;
#pragma unroll
  for (int t = 0; t < NIT; t++)
    gs += TSUM[((size_t)b * NIT + t) * 128 + j] * __expf(tm[t] - gm);
  const float invgs = 1.f / gs;
#pragma unroll
  for (int t = 0; t < NIT; t++)
    SCL[((size_t)b * NIT + t) * 128 + j] = __expf(tm[t] - gm) * invgs;
}

// ---------------------------------------------------------------------------
// Kernel 3: T = S2^T @ C; A = scaled P2T rows (f16), B = ChT rows (f16).
// Output stored transposed f16: TT[b,d,j].
// ---------------------------------------------------------------------------
__global__ __launch_bounds__(256) void k3_T(
    const _Float16* __restrict__ ChT, const _Float16* __restrict__ P2T,
    const float* __restrict__ SCL, _Float16* __restrict__ TT)
{
  const int b = blockIdx.x;
  const int tid = threadIdx.x, lane = tid & 31, wv = tid >> 5;
  const int h = lane >> 4, n = lane & 15, m = lane & 15;

  v8f acc[8] = {};
  const _Float16* prowbase = P2T + ((size_t)b * LQRY + 16 * wv + m) * LCTX;

  for (int it = 0; it < NIT; it++) {
    const _Float16 sclh = (_Float16)SCL[((size_t)b * NIT + it) * 128 + 16 * wv + m];
    const _Float16* prow = prowbase + it * 128;
    if (it + 1 < NIT) __builtin_prefetch(prow + 128, 0, 3);  // next K-chunk
#pragma unroll
    for (int ks = 0; ks < 4; ks++) {
      v16h a = load_afrag(prow + 32 * ks + 16 * h) * sclh;  // fold softmax rescale
#pragma unroll
      for (int nt = 0; nt < 8; nt++) {
        const _Float16* crow = ChT + ((size_t)b * DDIM + 16 * nt + n) * LCTX
                             + it * 128 + 32 * ks + 16 * h;
        v16h bf = *(const v16h*)crow;  // B[k=i][n=d] = C[i][d] = ChT[d][i]
        acc[nt] = wmma_f16(a, bf, acc[nt]);
      }
    }
  }

#pragma unroll
  for (int nt = 0; nt < 8; nt++) {
    _Float16* tb = TT + ((size_t)b * DDIM + 16 * nt + n) * LQRY + 16 * wv + 8 * h;
#pragma unroll
    for (int v = 0; v < 8; v++)
      tb[v] = (_Float16)acc[nt][v];  // TT[b][d][j], contiguous over v
  }
}

// ---------------------------------------------------------------------------
// Kernel 4: A_att = S1 @ Q, Bt = S1 @ T; out = [C, A, C*A, C*Bt].
// All B-fragments from pre-transposed f16 panels: zero LDS, zero conversions.
// ---------------------------------------------------------------------------
__global__ __launch_bounds__(256) void k4_out(
    const float* __restrict__ C, const _Float16* __restrict__ QhT,
    const _Float16* __restrict__ S1h, const _Float16* __restrict__ TT,
    float* __restrict__ out)
{
  const int b = blockIdx.y, it = blockIdx.x;
  const int i0 = it * 128;
  const int tid = threadIdx.x, lane = tid & 31, wv = tid >> 5;
  const int h = lane >> 4, n = lane & 15, m = lane & 15;

  const _Float16* s1row = S1h + ((size_t)b * LCTX + i0 + 16 * wv + m) * LQRY;
  v16h afr[4];
#pragma unroll
  for (int ks = 0; ks < 4; ks++) afr[ks] = load_afrag(s1row + 32 * ks + 16 * h);

  v8f accA[8] = {}, accB[8] = {};
#pragma unroll
  for (int ks = 0; ks < 4; ks++) {
#pragma unroll
    for (int nt = 0; nt < 8; nt++) {
      const size_t doff = ((size_t)b * DDIM + 16 * nt + n) * LQRY + 32 * ks + 16 * h;
      v16h bq = *(const v16h*)(QhT + doff);  // B[k=j][n=d] = Q[j][d] = QhT[d][j]
      v16h bt = *(const v16h*)(TT  + doff);
      accA[nt] = wmma_f16(afr[ks], bq, accA[nt]);
      accB[nt] = wmma_f16(afr[ks], bt, accB[nt]);
    }
  }

#pragma unroll
  for (int nt = 0; nt < 8; nt++) {
#pragma unroll
    for (int v = 0; v < 8; v++) {
      const int i = i0 + 16 * wv + v + 8 * h;
      const int d = 16 * nt + n;
      const float c  = C[((size_t)b * LCTX + i) * DDIM + d];
      const float av = accA[nt][v];
      const float bv = accB[nt][v];
      const size_t ob = ((size_t)b * LCTX + i) * (4 * DDIM);
      out[ob + d]            = c;
      out[ob + DDIM + d]     = av;
      out[ob + 2 * DDIM + d] = c * av;
      out[ob + 3 * DDIM + d] = c * bv;
    }
  }
}

// ---------------------------------------------------------------------------
extern "C" void kernel_launch(void* const* d_in, const int* in_sizes, int n_in,
                              void* d_out, int out_size, void* d_ws, size_t ws_size,
                              hipStream_t stream) {
  (void)in_sizes; (void)n_in; (void)out_size; (void)ws_size;
  const float* C     = (const float*)d_in[0];
  const float* Q     = (const float*)d_in[1];
  const float* cmask = (const float*)d_in[2];
  const float* qmask = (const float*)d_in[3];
  const float* w     = (const float*)d_in[4];
  float* out = (float*)d_out;

  char* ws = (char*)d_ws;
  size_t off = 0;
  _Float16* Cw3h = (_Float16*)(ws + off); off += (size_t)BATCH * LCTX * DDIM * 2;  // 16.8 MB
  _Float16* ChT  = (_Float16*)(ws + off); off += (size_t)BATCH * DDIM * LCTX * 2;  // 16.8 MB
  _Float16* Qh   = (_Float16*)(ws + off); off += (size_t)BATCH * LQRY * DDIM * 2;  // 1.05 MB
  _Float16* QhT  = (_Float16*)(ws + off); off += (size_t)BATCH * DDIM * LQRY * 2;  // 1.05 MB
  _Float16* S1h  = (_Float16*)(ws + off); off += (size_t)BATCH * LCTX * LQRY * 2;  // 16.8 MB
  _Float16* P2T  = (_Float16*)(ws + off); off += (size_t)BATCH * LQRY * LCTX * 2;  // 16.8 MB
  _Float16* TT   = (_Float16*)(ws + off); off += (size_t)BATCH * DDIM * LQRY * 2;  // 1.05 MB
  float* c1g  = (float*)(ws + off); off += (size_t)BATCH * LCTX * 4;
  float* q2g  = (float*)(ws + off); off += (size_t)BATCH * LQRY * 4;
  float* TMAX = (float*)(ws + off); off += (size_t)BATCH * NIT * 128 * 4;
  float* TSUM = (float*)(ws + off); off += (size_t)BATCH * NIT * 128 * 4;
  float* SCL  = (float*)(ws + off); off += (size_t)BATCH * NIT * 128 * 4;

  dim3 gt(NIT, BATCH);
  k0_prepC<<<gt, 256, 0, stream>>>(C, w, Cw3h, ChT, c1g);
  k0_prepQ<<<BATCH, 256, 0, stream>>>(Q, w, Qh, QhT, q2g);
  k1_score<<<gt, 256, 0, stream>>>(Cw3h, Qh, c1g, q2g, cmask, qmask,
                                   S1h, P2T, TMAX, TSUM);
  k2_colstats<<<BATCH, 128, 0, stream>>>(TMAX, TSUM, SCL);
  k3_T<<<BATCH, 256, 0, stream>>>(ChT, P2T, SCL, TT);
  k4_out<<<gt, 256, 0, stream>>>(C, QhT, S1h, TT, out);
}
